// ParallelMamba_67310727463150
// MI455X (gfx1250) — compile-verified, matrix operand order checked by reading, not run on hardware
//
#include <hip/hip_runtime.h>
#include <cmath>

// ---------------- problem constants ----------------
#define DMODEL   1024
#define D_INNER  2048
#define D_STATE  16
#define D_CONV   4
#define DT_RANK  64
#define NMAMBA   4
#define BATCH    2
#define LSEG     1024                  // L / NMAMBA
#define XPROJ_N  (DT_RANK + 2*D_STATE) // 96
#define XZ_W     (2*D_INNER)           // 4096

typedef __attribute__((ext_vector_type(2)))  float  v2f;
typedef __attribute__((ext_vector_type(4)))  float  v4f;
typedef __attribute__((ext_vector_type(8)))  float  v8f;
typedef __attribute__((ext_vector_type(4)))  __bf16 v4bf;
typedef __attribute__((ext_vector_type(8)))  __bf16 v8bf;
typedef __attribute__((ext_vector_type(16))) __bf16 v16bf;

#define SSTR 20   // fp32 LDS row stride (floats): float4-aligned, conflict-free
#define SH   40   // bf16 LDS row stride (halves): b128-aligned, conflict-free

// ---------------------------------------------------------------------------
// Split-bf16 (3-product) WMMA GEMM core:
//   C(BMxBN) += A(BMxK) * W(BNxK)^T  in fp32-class accuracy via
//   a_hi*w_hi + a_hi*w_lo + a_lo*w_hi with V_WMMA_F32_16X16X32_BF16.
// Loader reads fp32 from global and splits hi/lo into LDS on the fly.
// 256 threads = 8 waves; wave grid WAVES_M x WAVES_N; FM x FN frags per wave.
// ---------------------------------------------------------------------------
template<int WAVES_M, int WAVES_N, int FM, int FN, int K>
__device__ __forceinline__ void wmma_gemm_bf3_core(
    const float* __restrict__ Abase, int a_stride,
    const float* __restrict__ Wbase, int w_stride,
    __bf16* sAhi, __bf16* sAlo, __bf16* sWhi, __bf16* sWlo,
    v8f (&acc)[FM][FN])
{
  constexpr int BM = WAVES_M * FM * 16;
  constexpr int BN = WAVES_N * FN * 16;
  const int tid  = threadIdx.x;
  const int lane = tid & 31;
  const int wave = tid >> 5;
  const int wm   = wave % WAVES_M;
  const int wn   = wave / WAVES_M;
  const int lr   = lane & 15;   // row/col within fragment
  const int kh   = lane >> 4;   // K-half selector

  v8f zero = {};
  #pragma unroll
  for (int i = 0; i < FM; ++i)
    #pragma unroll
    for (int j = 0; j < FN; ++j) acc[i][j] = zero;

  for (int k0 = 0; k0 < K; k0 += 32) {
    __syncthreads();
    // stage BMx32 of A and BNx32 of W: load fp32, split to bf16 hi/lo in LDS
    for (int idx = tid; idx < (BM + BN) * 8; idx += WAVES_M * WAVES_N * 32) {
      int r = idx >> 3, c = idx & 7;          // row, float4-chunk within 32 cols
      const float* src;
      __bf16 *dhi, *dlo;
      if (r < BM) {
        src = Abase + (size_t)r * a_stride + k0 + c * 4;
        dhi = &sAhi[r * SH + c * 4];  dlo = &sAlo[r * SH + c * 4];
      } else {
        int rw = r - BM;
        src = Wbase + (size_t)rw * w_stride + k0 + c * 4;
        dhi = &sWhi[rw * SH + c * 4]; dlo = &sWlo[rw * SH + c * 4];
      }
      v4f v = *(const v4f*)src;
      v4bf hv, lv;
      #pragma unroll
      for (int e = 0; e < 4; ++e) {
        __bf16 h = (__bf16)v[e];
        hv[e] = h;
        lv[e] = (__bf16)(v[e] - (float)h);
      }
      *(v4bf*)dhi = hv;
      *(v4bf*)dlo = lv;
    }
    __syncthreads();

    // fragment gather: per lane K = [8*kh .. 8*kh+7] then [16+8*kh .. +7]
    v16bf ah[FM], al[FM], wh[FN], wl[FN];
    #pragma unroll
    for (int i = 0; i < FM; ++i) {
      const int ro = (wm * FM * 16 + i * 16 + lr) * SH;
      v8bf p0 = *(const v8bf*)&sAhi[ro + kh * 8];
      v8bf p1 = *(const v8bf*)&sAhi[ro + 16 + kh * 8];
      ah[i] = __builtin_shufflevector(p0, p1, 0,1,2,3,4,5,6,7,8,9,10,11,12,13,14,15);
      p0 = *(const v8bf*)&sAlo[ro + kh * 8];
      p1 = *(const v8bf*)&sAlo[ro + 16 + kh * 8];
      al[i] = __builtin_shufflevector(p0, p1, 0,1,2,3,4,5,6,7,8,9,10,11,12,13,14,15);
    }
    #pragma unroll
    for (int j = 0; j < FN; ++j) {
      const int ro = (wn * FN * 16 + j * 16 + lr) * SH;
      v8bf p0 = *(const v8bf*)&sWhi[ro + kh * 8];
      v8bf p1 = *(const v8bf*)&sWhi[ro + 16 + kh * 8];
      wh[j] = __builtin_shufflevector(p0, p1, 0,1,2,3,4,5,6,7,8,9,10,11,12,13,14,15);
      p0 = *(const v8bf*)&sWlo[ro + kh * 8];
      p1 = *(const v8bf*)&sWlo[ro + 16 + kh * 8];
      wl[j] = __builtin_shufflevector(p0, p1, 0,1,2,3,4,5,6,7,8,9,10,11,12,13,14,15);
    }

    #pragma unroll
    for (int i = 0; i < FM; ++i)
      #pragma unroll
      for (int j = 0; j < FN; ++j) {
        acc[i][j] = __builtin_amdgcn_wmma_f32_16x16x32_bf16(
            false, ah[i], false, wh[j], (short)0, acc[i][j], false, false);
        acc[i][j] = __builtin_amdgcn_wmma_f32_16x16x32_bf16(
            false, ah[i], false, wl[j], (short)0, acc[i][j], false, false);
        acc[i][j] = __builtin_amdgcn_wmma_f32_16x16x32_bf16(
            false, al[i], false, wh[j], (short)0, acc[i][j], false, false);
      }
  }
}

// ---------------------------------------------------------------------------
// Exact fp32 WMMA core (V_WMMA_F32_16X16X4_F32) — used for the tiny dt GEMM
// (K=64) that feeds exp() in the scan.
// ---------------------------------------------------------------------------
template<int WAVES_M, int WAVES_N, int FM, int FN, int K>
__device__ __forceinline__ void wmma_gemm_f32_core(
    const float* __restrict__ Abase, int a_stride,
    const float* __restrict__ Wbase, int w_stride,
    float* sA, float* sW, v8f (&acc)[FM][FN])
{
  constexpr int BM = WAVES_M * FM * 16;
  constexpr int BN = WAVES_N * FN * 16;
  const int tid  = threadIdx.x;
  const int lane = tid & 31;
  const int wave = tid >> 5;
  const int wm   = wave % WAVES_M;
  const int wn   = wave / WAVES_M;
  const int lr   = lane & 15;
  const int kh   = lane >> 4;

  v8f zero = {};
  #pragma unroll
  for (int i = 0; i < FM; ++i)
    #pragma unroll
    for (int j = 0; j < FN; ++j) acc[i][j] = zero;

  for (int k0 = 0; k0 < K; k0 += 16) {
    __syncthreads();
    for (int idx = tid; idx < (BM + BN) * 4; idx += WAVES_M * WAVES_N * 32) {
      int r = idx >> 2, q = idx & 3;
      if (r < BM) {
        *(v4f*)&sA[r * SSTR + q * 4] =
            *(const v4f*)&Abase[(size_t)r * a_stride + k0 + q * 4];
      } else {
        int rw = r - BM;
        *(v4f*)&sW[rw * SSTR + q * 4] =
            *(const v4f*)&Wbase[(size_t)rw * w_stride + k0 + q * 4];
      }
    }
    __syncthreads();

    #pragma unroll
    for (int kk = 0; kk < 4; ++kk) {
      v2f a[FM], b[FN];
      #pragma unroll
      for (int i = 0; i < FM; ++i)
        a[i] = *(const v2f*)&sA[(wm * FM * 16 + i * 16 + lr) * SSTR + kk * 4 + 2 * kh];
      #pragma unroll
      for (int j = 0; j < FN; ++j)
        b[j] = *(const v2f*)&sW[(wn * FN * 16 + j * 16 + lr) * SSTR + kk * 4 + 2 * kh];
      #pragma unroll
      for (int i = 0; i < FM; ++i)
        #pragma unroll
        for (int j = 0; j < FN; ++j)
          acc[i][j] = __builtin_amdgcn_wmma_f32_16x16x4_f32(
              false, a[i], false, b[j], (short)0, acc[i][j], false, false);
    }
  }
}

// ---------------------------------------------------------------------------
// 1) in_proj: xz[m][b][l][0:4096] = x_m @ in_w[m]^T    (M=2048,N=4096,K=1024)
//    x_m rows gather x[b, l*4+m, :]
// ---------------------------------------------------------------------------
__global__ void __launch_bounds__(256) k_inproj(const float* __restrict__ x,
                                                const float* __restrict__ in_w,
                                                float* __restrict__ xz)
{
  constexpr int FM = 2, FN = 4;   // BM = BN = 128
  __shared__ __bf16 smem[4 * 128 * SH];
  const int mt = blockIdx.x, nt = blockIdx.y, m = blockIdx.z;
  const int row0 = mt * 128;
  const int b  = row0 / LSEG;
  const int l0 = row0 % LSEG;

  const float* Abase = x + ((size_t)b * (LSEG * NMAMBA) + (size_t)l0 * NMAMBA + m) * DMODEL;
  const float* Wbase = in_w + ((size_t)m * XZ_W + (size_t)nt * 128) * DMODEL;

  v8f acc[FM][FN];
  wmma_gemm_bf3_core<4, 2, FM, FN, DMODEL>(
      Abase, NMAMBA * DMODEL, Wbase, DMODEL,
      smem, smem + 128 * SH, smem + 2 * 128 * SH, smem + 3 * 128 * SH, acc);

  const int lane = threadIdx.x & 31, wave = threadIdx.x >> 5;
  const int wm = wave % 4, wn = wave / 4;
  const int lr = lane & 15, kh = lane >> 4;
  float* Cbase = xz + ((size_t)m * BATCH * LSEG + row0) * XZ_W + (size_t)nt * 128;
  #pragma unroll
  for (int i = 0; i < FM; ++i)
    #pragma unroll
    for (int j = 0; j < FN; ++j)
      #pragma unroll
      for (int v = 0; v < 8; ++v)
        Cbase[(size_t)(wm * 32 + i * 16 + v + 8 * kh) * XZ_W + wn * 64 + j * 16 + lr] =
            acc[i][j][v];
}

// ---------------------------------------------------------------------------
// 2) depthwise causal conv (4 taps) + bias + SiLU : xz[:, :2048] -> xi
// ---------------------------------------------------------------------------
__global__ void __launch_bounds__(256) k_conv(const float* __restrict__ xz,
                                              const float* __restrict__ conv_w,
                                              const float* __restrict__ conv_b,
                                              float* __restrict__ xi)
{
  size_t g = (size_t)blockIdx.x * blockDim.x + threadIdx.x;
  if (g >= (size_t)NMAMBA * BATCH * LSEG * D_INNER) return;
  const int d = (int)(g % D_INNER);
  size_t t = g / D_INNER;
  const int l = (int)(t % LSEG);
  const size_t mb = t / LSEG;          // m*BATCH + b
  const int m = (int)(mb / BATCH);

  const float* w = conv_w + ((size_t)m * D_INNER + d) * D_CONV;
  float accv = conv_b[m * D_INNER + d];
  const float* src = xz + (mb * LSEG) * (size_t)XZ_W + d;
  #pragma unroll
  for (int j = 0; j < D_CONV; ++j) {
    int li = l - (D_CONV - 1) + j;
    if (li >= 0) accv += src[(size_t)li * XZ_W] * w[j];
  }
  xi[g] = accv * (1.f / (1.f + __expf(-accv)));  // SiLU
}

// ---------------------------------------------------------------------------
// 3) x_proj: proj = xi @ xproj_w^T   (M=2048, N=96, K=2048)
// ---------------------------------------------------------------------------
__global__ void __launch_bounds__(256) k_xproj(const float* __restrict__ xi,
                                               const float* __restrict__ xproj_w,
                                               float* __restrict__ proj)
{
  constexpr int FM = 2, FN = 3;                 // BM=128, BN=96
  __shared__ __bf16 smem[2 * 128 * SH + 2 * 96 * SH];
  const int mt = blockIdx.x, m = blockIdx.z;
  const int row0 = mt * 128;

  const float* Abase = xi + ((size_t)m * BATCH * LSEG + row0) * D_INNER;
  const float* Wbase = xproj_w + (size_t)m * XPROJ_N * D_INNER;

  v8f acc[FM][FN];
  wmma_gemm_bf3_core<4, 2, FM, FN, D_INNER>(
      Abase, D_INNER, Wbase, D_INNER,
      smem, smem + 128 * SH, smem + 2 * 128 * SH, smem + 2 * 128 * SH + 96 * SH, acc);

  const int lane = threadIdx.x & 31, wave = threadIdx.x >> 5;
  const int wm = wave % 4, wn = wave / 4;
  const int lr = lane & 15, kh = lane >> 4;
  float* Cbase = proj + ((size_t)m * BATCH * LSEG + row0) * XPROJ_N;
  #pragma unroll
  for (int i = 0; i < FM; ++i)
    #pragma unroll
    for (int j = 0; j < FN; ++j)
      #pragma unroll
      for (int v = 0; v < 8; ++v)
        Cbase[(size_t)(wm * 32 + i * 16 + v + 8 * kh) * XPROJ_N + wn * 48 + j * 16 + lr] =
            acc[i][j][v];
}

// ---------------------------------------------------------------------------
// 4) dt_proj + softplus: dt = softplus(proj[:, :64] @ dt_w^T + dt_b)
//    (M=2048, N=2048, K=64) — exact fp32 WMMA (feeds exp() in the scan)
// ---------------------------------------------------------------------------
__global__ void __launch_bounds__(256) k_dtproj(const float* __restrict__ proj,
                                                const float* __restrict__ dt_w,
                                                const float* __restrict__ dt_b,
                                                float* __restrict__ dt)
{
  constexpr int FM = 2, FN = 4;
  __shared__ float smem[(128 + 128) * SSTR];
  const int mt = blockIdx.x, nt = blockIdx.y, m = blockIdx.z;
  const int row0 = mt * 128;

  const float* Abase = proj + ((size_t)m * BATCH * LSEG + row0) * XPROJ_N;
  const float* Wbase = dt_w + ((size_t)m * D_INNER + (size_t)nt * 128) * DT_RANK;

  v8f acc[FM][FN];
  wmma_gemm_f32_core<4, 2, FM, FN, DT_RANK>(Abase, XPROJ_N, Wbase, DT_RANK,
                                            smem, smem + 128 * SSTR, acc);

  const int lane = threadIdx.x & 31, wave = threadIdx.x >> 5;
  const int wm = wave % 4, wn = wave / 4;
  const int lr = lane & 15, kh = lane >> 4;
  float* Cbase = dt + ((size_t)m * BATCH * LSEG + row0) * D_INNER + (size_t)nt * 128;
  #pragma unroll
  for (int i = 0; i < FM; ++i)
    #pragma unroll
    for (int j = 0; j < FN; ++j) {
      const int col = nt * 128 + wn * 64 + j * 16 + lr;
      const float bias = dt_b[m * D_INNER + col];
      #pragma unroll
      for (int v = 0; v < 8; ++v) {
        float val = acc[i][j][v] + bias;
        val = (val > 20.f) ? val : log1pf(__expf(val));   // softplus
        Cbase[(size_t)(wm * 32 + i * 16 + v + 8 * kh) * D_INNER + wn * 64 + j * 16 + lr] = val;
      }
    }
}

// ---------------------------------------------------------------------------
// 5) selective scan, fused with D-skip and SiLU(z) gate.
//    y written into the dead xi_raw half of xz (channels [0,2048)).
//    One thread per (m,b,d); state h[16] in registers; B/C broadcast via LDS.
// ---------------------------------------------------------------------------
__global__ void __launch_bounds__(256) k_scan(const float* __restrict__ dt,
                                              const float* __restrict__ proj,
                                              const float* __restrict__ xi,
                                              float* __restrict__ xz,
                                              const float* __restrict__ A_log,
                                              const float* __restrict__ D_skip)
{
  const int d = blockIdx.x * 256 + threadIdx.x;
  const int b = blockIdx.y, m = blockIdx.z;

  float An[D_STATE], h[D_STATE];
  #pragma unroll
  for (int s = 0; s < D_STATE; ++s) {
    An[s] = -__expf(A_log[((size_t)m * D_INNER + d) * D_STATE + s]);
    h[s] = 0.f;
  }
  const float dskip = D_skip[m * D_INNER + d];

  __shared__ float sBC[2 * D_STATE];
  const size_t projbase = ((size_t)m * BATCH + b) * LSEG * XPROJ_N;
  const size_t rowbase  = ((size_t)m * BATCH + b) * LSEG;

  for (int l = 0; l < LSEG; ++l) {
    __syncthreads();
    if (threadIdx.x < 2 * D_STATE)
      sBC[threadIdx.x] = proj[projbase + (size_t)l * XPROJ_N + DT_RANK + threadIdx.x];
    __syncthreads();

    const float dtv = dt[(rowbase + l) * (size_t)D_INNER + d];
    const float xv  = xi[(rowbase + l) * (size_t)D_INNER + d];
    const float dx  = dtv * xv;
    float y = 0.f;
    #pragma unroll
    for (int s = 0; s < D_STATE; ++s) {
      const float dA = __expf(dtv * An[s]);
      h[s] = dA * h[s] + dx * sBC[s];
      y += h[s] * sBC[D_STATE + s];
    }
    y += xv * dskip;
    const float zv = xz[(rowbase + l) * (size_t)XZ_W + D_INNER + d];
    y *= zv * (1.f / (1.f + __expf(-zv)));           // * silu(z)
    xz[(rowbase + l) * (size_t)XZ_W + d] = y;        // overwrite dead xi_raw slot
  }
}

// ---------------------------------------------------------------------------
// 6) out_proj: out[b, m*1024+l, :] = y @ out_w[m]^T  (M=2048, N=1024, K=2048)
// ---------------------------------------------------------------------------
__global__ void __launch_bounds__(256) k_outproj(const float* __restrict__ xz,
                                                 const float* __restrict__ out_w,
                                                 float* __restrict__ out)
{
  constexpr int FM = 2, FN = 4;
  __shared__ __bf16 smem[4 * 128 * SH];
  const int mt = blockIdx.x, nt = blockIdx.y, m = blockIdx.z;
  const int row0 = mt * 128;
  const int b  = row0 / LSEG;
  const int l0 = row0 % LSEG;

  const float* Abase = xz + ((size_t)m * BATCH * LSEG + row0) * XZ_W;   // y half
  const float* Wbase = out_w + ((size_t)m * DMODEL + (size_t)nt * 128) * D_INNER;

  v8f acc[FM][FN];
  wmma_gemm_bf3_core<4, 2, FM, FN, D_INNER>(
      Abase, XZ_W, Wbase, D_INNER,
      smem, smem + 128 * SH, smem + 2 * 128 * SH, smem + 3 * 128 * SH, acc);

  const int lane = threadIdx.x & 31, wave = threadIdx.x >> 5;
  const int wm = wave % 4, wn = wave / 4;
  const int lr = lane & 15, kh = lane >> 4;
  float* Cbase = out + ((size_t)b * (NMAMBA * LSEG) + (size_t)m * LSEG + l0) * DMODEL
               + (size_t)nt * 128;
  #pragma unroll
  for (int i = 0; i < FM; ++i)
    #pragma unroll
    for (int j = 0; j < FN; ++j)
      #pragma unroll
      for (int v = 0; v < 8; ++v)
        Cbase[(size_t)(wm * 32 + i * 16 + v + 8 * kh) * DMODEL + wn * 64 + j * 16 + lr] =
            acc[i][j][v];
}

// ---------------------------------------------------------------------------
extern "C" void kernel_launch(void* const* d_in, const int* in_sizes, int n_in,
                              void* d_out, int out_size, void* d_ws, size_t ws_size,
                              hipStream_t stream)
{
  const float* x       = (const float*)d_in[0];
  const float* in_w    = (const float*)d_in[1];
  const float* conv_w  = (const float*)d_in[2];
  const float* conv_b  = (const float*)d_in[3];
  const float* xproj_w = (const float*)d_in[4];
  const float* dt_w    = (const float*)d_in[5];
  const float* dt_b    = (const float*)d_in[6];
  const float* A_log   = (const float*)d_in[7];
  const float* D_skip  = (const float*)d_in[8];
  const float* out_w   = (const float*)d_in[9];
  float* out = (float*)d_out;

  // workspace layout (floats)
  float* ws   = (float*)d_ws;
  float* xz   = ws;                                                   // 4*2*1024*4096
  float* xi   = xz   + (size_t)NMAMBA * BATCH * LSEG * XZ_W;          // 4*2*1024*2048
  float* proj = xi   + (size_t)NMAMBA * BATCH * LSEG * D_INNER;       // 4*2*1024*96
  float* dt   = proj + (size_t)NMAMBA * BATCH * LSEG * XPROJ_N;       // 4*2*1024*2048

  const dim3 blk(256);
  k_inproj <<<dim3(16, 32, NMAMBA), blk, 0, stream>>>(x, in_w, xz);

  const size_t convN = (size_t)NMAMBA * BATCH * LSEG * D_INNER;
  k_conv   <<<dim3((unsigned)((convN + 255) / 256)), blk, 0, stream>>>(xz, conv_w, conv_b, xi);

  k_xproj  <<<dim3(16, 1,  NMAMBA), blk, 0, stream>>>(xi, xproj_w, proj);
  k_dtproj <<<dim3(16, 16, NMAMBA), blk, 0, stream>>>(proj, dt_w, dt_b, dt);
  k_scan   <<<dim3(D_INNER / 256, BATCH, NMAMBA), blk, 0, stream>>>(dt, proj, xi, xz, A_log, D_skip);
  k_outproj<<<dim3(16, 8,  NMAMBA), blk, 0, stream>>>(xz, out_w, out);
}
